// RetentiveAttention_22952305230023
// MI455X (gfx1250) — matrix operand order repaired and checked by reading, not compiled, not run on hardware
//
#include <hip/hip_runtime.h>

// ---------------------------------------------------------------------------
// CDNA5 (gfx1250) retentive attention, bf16 WMMA pipeline, wave32.
// Async LDS staging (GLOBAL_LOAD_ASYNC_TO_LDS_B128) + double buffering.
// ---------------------------------------------------------------------------

typedef __bf16 bf16_t;
typedef __attribute__((ext_vector_type(16))) __bf16 v16bf;
typedef __attribute__((ext_vector_type(8)))  __bf16 v8bf;
typedef __attribute__((ext_vector_type(8)))  float  v8f;
typedef __attribute__((ext_vector_type(4)))  float  v4f;

#define D_MODEL   1024
#define NUM_HEADS 16
#define HEAD_DIM  64
#define BATCH     2
#define SEQ       2048
#define M_TOK     (BATCH * SEQ)   // 4096

// D = A(16x32 bf16) x B(32x16 bf16) + C(16x16 f32)
__device__ __forceinline__ v8f wmma_bf16(v16bf a, v16bf b, v8f c) {
  return __builtin_amdgcn_wmma_f32_16x16x32_bf16(
      /*neg_a=*/false, a, /*neg_b=*/false, b,
      /*c_mod=*/(short)0, c, /*reuse_a=*/false, /*reuse_b=*/false);
}

__device__ __forceinline__ v16bf mk_v16(v8bf lo, v8bf hi) {
  v16bf r;
#pragma unroll
  for (int i = 0; i < 8; ++i) { r[i] = lo[i]; r[i + 8] = hi[i]; }
  return r;
}

// A-matrix 16x32 bf16 fragment: lane m = lane%16, khalf = lane/16.
// elements 0..7  = K[16*khalf .. 16*khalf+7]
// elements 8..15 = K[16*khalf+8 .. 16*khalf+15]
__device__ __forceinline__ v16bf load_a_frag(const bf16_t* row, int khalf) {
  v8bf lo = *(const v8bf*)(row + 16 * khalf);
  v8bf hi = *(const v8bf*)(row + 16 * khalf + 8);
  return mk_v16(lo, hi);
}

// 16B async copy global -> LDS (ASYNCcnt-tracked, no VGPR round trip).
// vdst = 32-bit LDS byte address (low bits of generic shared pointer),
// vaddr = 64-bit global address.
__device__ __forceinline__ void async_ld16(void* lds, const void* gptr) {
  unsigned lds_off = (unsigned)(size_t)lds;
  asm volatile("global_load_async_to_lds_b128 %0, %1, off"
               :
               : "v"(lds_off), "v"(gptr)
               : "memory");
}

__device__ __forceinline__ void wait_async0() {
  asm volatile("s_wait_asynccnt 0x0" ::: "memory");
}

// ---------------------------------------------------------------------------
// f32 -> bf16 convert (4-wide)
// ---------------------------------------------------------------------------
__global__ void cvt_f32_bf16(const float* __restrict__ in,
                             bf16_t* __restrict__ out, int n4) {
  int i = blockIdx.x * blockDim.x + threadIdx.x;
  if (i < n4) {
    v4f v = *(const v4f*)(in + 4 * (size_t)i);
    bf16_t o[4];
#pragma unroll
    for (int j = 0; j < 4; ++j) o[j] = (bf16_t)v[j];
    *(uint2*)(out + 4 * (size_t)i) = *(uint2*)o;  // 8 bytes
  }
}

// ---------------------------------------------------------------------------
// GEMM: C[m,n] = sum_k A[m,k] * W[n,k] + bias[n]
// A: [M,K] bf16 row-major, W: [N,K] bf16 row-major (torch Linear layout).
// Block tile 128x128, 8 waves, each wave 32x64 (2x4 WMMA tiles), K step 32.
// Double-buffered async LDS staging; one barrier per K step.
// MODE 0: bf16 row-major output [M,N]
// MODE 1: bf16 per-head transposed V output [B][H][HEAD_DIM][SEQ]
// MODE 2: f32 row-major output [M,N]
// ---------------------------------------------------------------------------
template <int MODE>
__global__ __launch_bounds__(256)
void gemm_xwT(const bf16_t* __restrict__ A, const bf16_t* __restrict__ W,
              const float* __restrict__ bias, bf16_t* __restrict__ outb,
              float* __restrict__ outf, int M, int N, int K) {
  __shared__ __align__(64) bf16_t ldsA[2][128 * 32];
  __shared__ __align__(64) bf16_t ldsB[2][128 * 32];

  const int tid  = threadIdx.x;
  const int lane = tid & 31;
  const int wave = tid >> 5;
  const int wr   = wave & 3;   // M strip (32 rows)
  const int wc   = wave >> 2;  // N strip (64 cols)
  const int col  = lane & 15;  // A: row m within tile; B/C: column n
  const int hi   = lane >> 4;  // A: K half select; C: row-half select
  const int m0   = blockIdx.x * 128;
  const int n0   = blockIdx.y * 128;

  v8f acc[2][4];
#pragma unroll
  for (int mt = 0; mt < 2; ++mt)
#pragma unroll
    for (int nt = 0; nt < 4; ++nt) acc[mt][nt] = (v8f)0.0f;

  // async stage of one 128x32 A tile + 128x32 W tile into buffer `buf`
  auto stageAB = [&](int buf, int k0) {
#pragma unroll
    for (int s = 0; s < 2; ++s) {
      int c  = tid + s * 256;
      int r  = c >> 2;
      int kk = (c & 3) * 8;
      async_ld16(&ldsA[buf][r * 32 + kk],
                 &A[(size_t)(m0 + r) * K + k0 + kk]);
      async_ld16(&ldsB[buf][r * 32 + kk],
                 &W[(size_t)(n0 + r) * K + k0 + kk]);
    }
  };

  stageAB(0, 0);
  wait_async0();
  __syncthreads();

  for (int k0 = 0; k0 < K; k0 += 32) {
    const int cur = (k0 >> 5) & 1;
    if (k0 + 32 < K) stageAB(cur ^ 1, k0 + 32);  // overlap copy with WMMA

    v16bf af[2];
#pragma unroll
    for (int mt = 0; mt < 2; ++mt)
      af[mt] = load_a_frag(&ldsA[cur][(wr * 32 + mt * 16 + col) * 32], hi);

#pragma unroll
    for (int nt = 0; nt < 4; ++nt) {
      // B fragment: lane col n, 16 contiguous K at 16*hi
      v16bf bf =
          *(const v16bf*)&ldsB[cur][(wc * 64 + nt * 16 + col) * 32 + 16 * hi];
#pragma unroll
      for (int mt = 0; mt < 2; ++mt)
        acc[mt][nt] = wmma_bf16(af[mt], bf, acc[mt][nt]);
    }

    wait_async0();
    __syncthreads();
  }

  // epilogue: C layout -> lane holds column n, rows m = 8*hi + j
#pragma unroll
  for (int mt = 0; mt < 2; ++mt) {
#pragma unroll
    for (int nt = 0; nt < 4; ++nt) {
      const int ng   = n0 + wc * 64 + nt * 16 + col;
      const float bv = bias[ng];
      const int mg0  = m0 + wr * 32 + mt * 16 + 8 * hi;
      if (MODE == 0) {
#pragma unroll
        for (int j = 0; j < 8; ++j)
          outb[(size_t)(mg0 + j) * N + ng] = (bf16_t)(acc[mt][nt][j] + bv);
      } else if (MODE == 1) {
        const int b = mg0 / SEQ;
        const int l = mg0 % SEQ;
        const int h = ng >> 6;
        const int d = ng & 63;
        bf16_t tmp[8];
#pragma unroll
        for (int j = 0; j < 8; ++j) tmp[j] = (bf16_t)(acc[mt][nt][j] + bv);
        *(v8bf*)&outb[((size_t)(b * NUM_HEADS + h) * HEAD_DIM + d) * SEQ + l] =
            *(v8bf*)tmp;
      } else {
#pragma unroll
        for (int j = 0; j < 8; ++j)
          outf[(size_t)(mg0 + j) * N + ng] = acc[mt][nt][j] + bv;
      }
    }
  }
}

// ---------------------------------------------------------------------------
// Fused retention attention. Q,K: [M_TOK, D_MODEL] bf16 token-major.
// Vt: [B][H][HEAD_DIM][SEQ] bf16.  Ab out: [M_TOK, D_MODEL] bf16.
// Block: 8 waves = 128 queries of one (b,h); keys streamed in chunks of 32
// with double-buffered async LDS staging.
// out[q,:] = gamma_h^q * sum_{k<=q} (q.kT)[q,k] * v[k,:]
// ---------------------------------------------------------------------------
__global__ __launch_bounds__(256)
void retention_attn(const bf16_t* __restrict__ Q, const bf16_t* __restrict__ Kb,
                    const bf16_t* __restrict__ Vt,
                    const float* __restrict__ gamma,
                    bf16_t* __restrict__ Ab) {
  __shared__ __align__(64) bf16_t ldsK[2][32 * 64];  // 32 keys x 64 d, x2
  __shared__ __align__(64) bf16_t ldsS[8][16 * 32];  // per-wave score relayout

  const int tid  = threadIdx.x;
  const int lane = tid & 31;
  const int wave = tid >> 5;
  const int col  = lane & 15;
  const int hi   = lane >> 4;
  const int qb0  = blockIdx.x * 128;  // query block within sequence
  const int bh   = blockIdx.y;
  const int b    = bh >> 4;
  const int h    = bh & 15;
  const int qw0  = qb0 + wave * 16;   // this wave's 16 queries

  // Q A-fragments for both 32-wide d steps (reused across all key chunks)
  const bf16_t* qrow =
      Q + (size_t)(b * SEQ + qw0 + col) * D_MODEL + h * HEAD_DIM;
  v16bf qf[2];
#pragma unroll
  for (int ks = 0; ks < 2; ++ks) qf[ks] = load_a_frag(qrow + ks * 32, hi);

  v8f oacc[4];
#pragma unroll
  for (int dt = 0; dt < 4; ++dt) oacc[dt] = (v8f)0.0f;

  // async stage of one 32x64 K chunk (one 16B async copy per thread)
  auto stageK = [&](int buf, int kc) {
    int r  = tid >> 3;
    int cc = (tid & 7) * 8;
    async_ld16(&ldsK[buf][r * 64 + cc],
               &Kb[(size_t)(b * SEQ + kc * 32 + r) * D_MODEL +
                   h * HEAD_DIM + cc]);
  };

  const int nchunk = (qb0 + 128) / 32;  // causal bound for whole block
  stageK(0, 0);
  wait_async0();
  __syncthreads();

  for (int kc = 0; kc < nchunk; ++kc) {
    const int cur = kc & 1;
    if (kc + 1 < nchunk) stageK(cur ^ 1, kc + 1);  // overlap with WMMA

    if (kc * 32 <= qw0 + 15) {  // this wave still has causal work here
      // scores: two 16-key tiles, K-dim = d (64 = 2 x 32)
#pragma unroll
      for (int t = 0; t < 2; ++t) {
        v8f s = (v8f)0.0f;
#pragma unroll
        for (int ks = 0; ks < 2; ++ks) {
          v16bf kf = *(const v16bf*)&ldsK[cur][(t * 16 + col) * 64 +
                                               ks * 32 + 16 * hi];
          s = wmma_bf16(qf[ks], kf, s);
        }
        // causal mask + relayout C(f32) -> A(bf16) through per-wave LDS
#pragma unroll
        for (int j = 0; j < 8; ++j) {
          int qg = qw0 + 8 * hi + j;
          int kg = kc * 32 + t * 16 + col;
          float sv = (kg <= qg) ? s[j] : 0.0f;
          ldsS[wave][(8 * hi + j) * 32 + t * 16 + col] = (bf16_t)sv;
        }
      }
      v16bf sf = load_a_frag(&ldsS[wave][col * 32], hi);
      // out += s(16x32) x V(32 keys x 64 d); Vt fragments contiguous in global
#pragma unroll
      for (int dt = 0; dt < 4; ++dt) {
        v16bf vf = *(const v16bf*)&Vt[
            ((size_t)(b * NUM_HEADS + h) * HEAD_DIM + dt * 16 + col) * SEQ +
            kc * 32 + 16 * hi];
        oacc[dt] = wmma_bf16(sf, vf, oacc[dt]);
      }
    }

    wait_async0();
    __syncthreads();
  }

  // per-query retention decay gamma^q, then bf16 store token-major
  // gamma^q = exp2(q * log2(gamma)); v_log_f32 / v_exp_f32 are base-2.
  const float lg = __builtin_amdgcn_logf(gamma[h]);
#pragma unroll
  for (int dt = 0; dt < 4; ++dt) {
#pragma unroll
    for (int j = 0; j < 8; ++j) {
      int qg = qw0 + 8 * hi + j;
      float scale = __builtin_amdgcn_exp2f((float)qg * lg);
      Ab[(size_t)(b * SEQ + qg) * D_MODEL + h * HEAD_DIM + dt * 16 + col] =
          (bf16_t)(oacc[dt][j] * scale);
    }
  }
}

// ---------------------------------------------------------------------------
// launcher
// ---------------------------------------------------------------------------
extern "C" void kernel_launch(void* const* d_in, const int* in_sizes, int n_in,
                              void* d_out, int out_size, void* d_ws,
                              size_t ws_size, hipStream_t stream) {
  const float* x     = (const float*)d_in[0];
  const float* Wq    = (const float*)d_in[1];
  const float* bq    = (const float*)d_in[2];
  const float* Wk    = (const float*)d_in[3];
  const float* bk    = (const float*)d_in[4];
  const float* Wv    = (const float*)d_in[5];
  const float* bv    = (const float*)d_in[6];
  const float* Wo    = (const float*)d_in[7];
  const float* bo    = (const float*)d_in[8];
  const float* gamma = (const float*)d_in[9];
  float* out = (float*)d_out;

  const size_t NX = (size_t)M_TOK * D_MODEL;    // 4,194,304
  const size_t NW = (size_t)D_MODEL * D_MODEL;  // 1,048,576

  char* p = (char*)d_ws;
  bf16_t* xb  = (bf16_t*)p; p += NX * 2;
  bf16_t* Wqb = (bf16_t*)p; p += NW * 2;
  bf16_t* Wkb = (bf16_t*)p; p += NW * 2;
  bf16_t* Wvb = (bf16_t*)p; p += NW * 2;
  bf16_t* Wob = (bf16_t*)p; p += NW * 2;
  bf16_t* Qb  = (bf16_t*)p; p += NX * 2;
  bf16_t* Kb  = (bf16_t*)p; p += NX * 2;
  bf16_t* Vtb = (bf16_t*)p; p += NX * 2;  // [B][H][HEAD_DIM][SEQ]
  bf16_t* Ab  = (bf16_t*)p; p += NX * 2;

  // 1) converts
  {
    int n4 = (int)(NX / 4);
    cvt_f32_bf16<<<(n4 + 255) / 256, 256, 0, stream>>>(x, xb, n4);
    int w4 = (int)(NW / 4);
    cvt_f32_bf16<<<(w4 + 255) / 256, 256, 0, stream>>>(Wq, Wqb, w4);
    cvt_f32_bf16<<<(w4 + 255) / 256, 256, 0, stream>>>(Wk, Wkb, w4);
    cvt_f32_bf16<<<(w4 + 255) / 256, 256, 0, stream>>>(Wv, Wvb, w4);
    cvt_f32_bf16<<<(w4 + 255) / 256, 256, 0, stream>>>(Wo, Wob, w4);
  }

  // 2) projections
  dim3 gg(M_TOK / 128, D_MODEL / 128);  // (32, 8)
  gemm_xwT<0><<<gg, 256, 0, stream>>>(xb, Wqb, bq, Qb, nullptr,
                                      M_TOK, D_MODEL, D_MODEL);
  gemm_xwT<0><<<gg, 256, 0, stream>>>(xb, Wkb, bk, Kb, nullptr,
                                      M_TOK, D_MODEL, D_MODEL);
  gemm_xwT<1><<<gg, 256, 0, stream>>>(xb, Wvb, bv, Vtb, nullptr,
                                      M_TOK, D_MODEL, D_MODEL);

  // 3) fused retention attention
  dim3 ga(SEQ / 128, BATCH * NUM_HEADS);  // (16, 32)
  retention_attn<<<ga, 256, 0, stream>>>(Qb, Kb, Vtb, gamma, Ab);

  // 4) output projection (f32 epilogue to d_out)
  gemm_xwT<2><<<gg, 256, 0, stream>>>(Ab, Wob, bo, nullptr, out,
                                      M_TOK, D_MODEL, D_MODEL);
}
